// TransformerBlock_42460046688341
// MI455X (gfx1250) — matrix operand order repaired and checked by reading, not compiled, not run on hardware
//
#include <hip/hip_runtime.h>

// ---------------------------------------------------------------------------
// TransformerBlock for MI455X (gfx1250): bf16 WMMA everywhere.
// B=8, T=512, D=2048, H=32, hd=64, FFN=8192.
// GEMM: 64x128 block tile, 16x64 wave tile, K-step 64 -> 8 WMMA per LDS stage.
// ---------------------------------------------------------------------------

typedef __attribute__((ext_vector_type(16))) __bf16 v16bf;
typedef __attribute__((ext_vector_type(8)))  float  v8f;

union Frag {
    v16bf v;
    uint4 q[2];
};

__device__ inline unsigned short f2bf(float f) {
    unsigned int u = __float_as_uint(f);
    unsigned int r = (u + 0x7FFFu + ((u >> 16) & 1u)) >> 16;   // round-to-nearest-even
    return (unsigned short)r;
}
__device__ inline float bf2f(unsigned short u) {
    return __uint_as_float(((unsigned int)u) << 16);
}
__device__ inline v8f vzero8() {
    v8f z;
    for (int i = 0; i < 8; ++i) z[i] = 0.0f;
    return z;
}
__device__ inline v8f wmma_bf16(const Frag& a, const Frag& b, v8f c) {
    return __builtin_amdgcn_wmma_f32_16x16x32_bf16(false, a.v, false, b.v,
                                                   (short)0, c, false, false);
}

// ---------------------------------------------------------------------------
// RMSNorm: one block per row of 2048, output bf16 (ushort storage).
// ---------------------------------------------------------------------------
__global__ void rmsnorm_kernel(const float* __restrict__ x,
                               const float* __restrict__ w,
                               unsigned short* __restrict__ out) {
    const int row = blockIdx.x;
    const int tid = threadIdx.x;
    const float* xr = x + (size_t)row * 2048;
    float v[8];
    float s = 0.0f;
    for (int j = 0; j < 8; ++j) {
        v[j] = xr[tid + j * 256];
        s += v[j] * v[j];
    }
    __shared__ float red[256];
    red[tid] = s;
    __syncthreads();
    for (int off = 128; off > 0; off >>= 1) {
        if (tid < off) red[tid] += red[tid + off];
        __syncthreads();
    }
    const float rs = rsqrtf(red[0] * (1.0f / 2048.0f) + 1e-6f);
    unsigned short* o = out + (size_t)row * 2048;
    for (int j = 0; j < 8; ++j) {
        const int c = tid + j * 256;
        o[c] = f2bf(v[j] * rs * w[c]);
    }
}

// ---------------------------------------------------------------------------
// GEMM: C[M,N] = A[M,K](bf16) * W[N,K](fp32->bf16)^T, WMMA bf16, fp32 acc.
// Block = 256 threads = 8 waves.  Block tile 64(M) x 128(N), K-step 64.
// Wave tile 16(M) x 64(N): one A fragment reused across 4 B fragments,
// 8 WMMA per LDS stage per wave.  Epilogue fused per MODE.
// ---------------------------------------------------------------------------
#define GM_F32        0  // fp32 [M,N] store
#define GM_HEADS_DUAL 1  // fp32 + bf16 stores in [B,H,T,hd] layout (for V)
#define GM_RESID      2  // fp32 [M,N] store with residual add
#define GM_SILU       3  // bf16 silu(c)
#define GM_MUL        4  // bf16 c * aux[m,n]

template <int MODE>
__global__ void gemm_bf16_kernel(const unsigned short* __restrict__ A,
                                 const float* __restrict__ W,
                                 int M, int N, int K,
                                 float* __restrict__ outF,
                                 unsigned short* __restrict__ outB,
                                 const float* __restrict__ resid,
                                 const unsigned short* __restrict__ aux) {
    constexpr int BM = 64, BN = 128, BK = 64;
    constexpr int LDK = BK + 8;                 // 72 halves = 144B rows (16B aligned)
    __shared__ unsigned short As[BM * LDK];     //  9.2 KB
    __shared__ unsigned short Bs[BN * LDK];     // 18.4 KB

    const int tid = threadIdx.x;
    const int wv  = tid >> 5;
    const int ln  = tid & 31;
    const int hi  = ln >> 4;
    const int lc  = ln & 15;
    const int blockM = blockIdx.y * BM;
    const int blockN = blockIdx.x * BN;
    const int mrow = (wv & 3) * 16;             // 4 wave-rows
    const int ncol = (wv >> 2) * 64;            // 2 wave-cols of 64

    v8f acc[4];
    for (int c = 0; c < 4; ++c) acc[c] = vzero8();

    // Loaders: A tile 64x64 bf16 (16 halves/thread), W tile 128x64 fp32->bf16
    // (32 floats/thread).
    const int ar = tid >> 2, ac = (tid & 3) * 16;
    const int br = tid >> 1, bc = (tid & 1) * 32;
    const int klo = hi ? 8 : 0;
    const int kbn = hi ? 8 : 0;

    for (int k0 = 0; k0 < K; k0 += BK) {
        // Stage A tile (bf16, straight copy)
        {
            const uint4* src = reinterpret_cast<const uint4*>(
                A + (size_t)(blockM + ar) * K + k0 + ac);
            uint4* dst = reinterpret_cast<uint4*>(&As[ar * LDK + ac]);
            dst[0] = src[0];
            dst[1] = src[1];
        }
        // Stage W tile (fp32 -> bf16 convert)
        {
            const float4* src = reinterpret_cast<const float4*>(
                W + (size_t)(blockN + br) * K + k0 + bc);
            for (int g = 0; g < 4; ++g) {
                const float4 f0 = src[2 * g];
                const float4 f1 = src[2 * g + 1];
                unsigned short t8[8] = {f2bf(f0.x), f2bf(f0.y), f2bf(f0.z), f2bf(f0.w),
                                        f2bf(f1.x), f2bf(f1.y), f2bf(f1.z), f2bf(f1.w)};
                *reinterpret_cast<uint4*>(&Bs[br * LDK + bc + 8 * g]) =
                    *reinterpret_cast<uint4*>(t8);
            }
        }
        if (k0 + BK < K)
            __builtin_prefetch(W + (size_t)(blockN + br) * K + k0 + BK + bc, 0, 0);
        __syncthreads();

        // 2 K-substeps of 32; A fragment reused across the 4 N sub-tiles.
        for (int ks = 0; ks < 2; ++ks) {
            Frag a;
            const unsigned short* arow = &As[(mrow + lc) * LDK + 32 * ks];
            a.q[0] = *reinterpret_cast<const uint4*>(arow + klo);
            a.q[1] = *reinterpret_cast<const uint4*>(arow + 16 + klo);
            for (int c = 0; c < 4; ++c) {
                Frag b;
                const unsigned short* brow =
                    &Bs[(ncol + 16 * c + lc) * LDK + 32 * ks + kbn];
                b.q[0] = *reinterpret_cast<const uint4*>(brow);
                b.q[1] = *reinterpret_cast<const uint4*>(brow + 16);
                acc[c] = wmma_bf16(a, b, acc[c]);
            }
        }
        __syncthreads();
    }

    // Epilogue: C layout = VGPR i holds M = i + 8*hi, N = lc (per sub-tile).
    for (int c = 0; c < 4; ++c) {
        const int n = blockN + ncol + 16 * c + lc;
        for (int i = 0; i < 8; ++i) {
            const int m = blockM + mrow + i + 8 * hi;
            const float cv = acc[c][i];
            if constexpr (MODE == GM_F32) {
                outF[(size_t)m * N + n] = cv;
            } else if constexpr (MODE == GM_HEADS_DUAL) {
                const int b8 = m >> 9, t = m & 511;   // T = 512
                const int h = n >> 6, d = n & 63;     // hd = 64
                const size_t oi = (((size_t)(b8 * 32 + h)) * 512 + t) * 64 + d;
                outF[oi] = cv;
                outB[oi] = f2bf(cv);
            } else if constexpr (MODE == GM_RESID) {
                const size_t oi = (size_t)m * N + n;
                outF[oi] = cv + resid[oi];
            } else if constexpr (MODE == GM_SILU) {
                const float s = cv / (1.0f + __expf(-cv));
                outB[(size_t)m * N + n] = f2bf(s);
            } else if constexpr (MODE == GM_MUL) {
                const size_t oi = (size_t)m * N + n;
                outB[oi] = f2bf(cv * bf2f(aux[oi]));
            }
        }
    }
}

// ---------------------------------------------------------------------------
// RoPE: reads fp32 Q/K ([B,T,D]), writes bf16 Q/K in heads layout [B,H,T,hd]
// and roped K (fp32) into d_out's new_k region.
// ---------------------------------------------------------------------------
__global__ void rope_kernel(const float* __restrict__ qpre,
                            const float* __restrict__ kpre,
                            const int* __restrict__ posp,
                            unsigned short* __restrict__ qbf,
                            unsigned short* __restrict__ kbf,
                            float* __restrict__ kout) {
    const int idx = blockIdx.x * 256 + threadIdx.x;  // ((b*H+h)*T+t)*32 + i
    const int i = idx & 31;
    const int r = idx >> 5;        // (b*H+h)*T + t
    const int t = r & 511;
    const int bh = r >> 9;
    const int h = bh & 31;
    const int b = bh >> 5;
    // inv = theta^(-2i/64)
    const float inv = __expf(-(float)(2 * i) * (1.0f / 64.0f) * 9.210340371976184f);
    const float pos = (float)(posp[0] + t);
    float sn, cs;
    __sincosf(pos * inv, &sn, &cs);
    const size_t src = ((size_t)(b * 512 + t)) * 2048 + h * 64 + 2 * i;
    const float q0 = qpre[src], q1 = qpre[src + 1];
    const float k0 = kpre[src], k1 = kpre[src + 1];
    const float qo0 = q0 * cs - q1 * sn, qo1 = q0 * sn + q1 * cs;
    const float ko0 = k0 * cs - k1 * sn, ko1 = k0 * sn + k1 * cs;
    const size_t dst = (size_t)r * 64 + 2 * i;
    qbf[dst] = f2bf(qo0);  qbf[dst + 1] = f2bf(qo1);
    kbf[dst] = f2bf(ko0);  kbf[dst + 1] = f2bf(ko1);
    kout[dst] = ko0;       kout[dst + 1] = ko1;
}

// ---------------------------------------------------------------------------
// Flash attention (causal). Grid (T/128, B*H); 256 threads = 8 waves;
// wave w owns 16 query rows. K/V staged in LDS per 32-key step; scores via
// WMMA (2 N-halves), online softmax with width-16 shuffles, P transposed
// through per-wave LDS, V kept transposed in LDS so P*V is 4 more WMMAs.
// Output written bf16 in [B,T,D] layout (ready for the wo GEMM).
// ---------------------------------------------------------------------------
__global__ void attn_kernel(const unsigned short* __restrict__ qb,
                            const unsigned short* __restrict__ kb,
                            const unsigned short* __restrict__ vb,
                            unsigned short* __restrict__ attn) {
    constexpr int LKT = 72, LVT = 40, LPW = 40;
    __shared__ unsigned short Kt[32 * LKT];       // [kt][d]
    __shared__ unsigned short Vt[64 * LVT];       // [d][kt]  (transposed)
    __shared__ unsigned short Pw[8 * 16 * LPW];   // per-wave P scratch

    const int tid = threadIdx.x;
    const int wv  = tid >> 5;
    const int ln  = tid & 31;
    const int hi  = ln >> 4;
    const int lc  = ln & 15;
    const int bh  = blockIdx.y;           // b*32 + h
    const int qblk = blockIdx.x;          // 128 query rows per block
    const int qbase = qblk * 128 + wv * 16;
    const size_t headOff = (size_t)bh * 512 * 64;

    // Preload Q fragments (d = 0..63 -> two K=32 fragments).
    Frag qf[2];
    {
        const int row = qbase + lc;
        const int klo = hi ? 8 : 0;
        const unsigned short* qrow = qb + headOff + (size_t)row * 64;
        for (int j = 0; j < 2; ++j) {
            qf[j].q[0] = *reinterpret_cast<const uint4*>(qrow + 32 * j + klo);
            qf[j].q[1] = *reinterpret_cast<const uint4*>(qrow + 32 * j + 16 + klo);
        }
    }

    v8f o[4];
    for (int c = 0; c < 4; ++c) o[c] = vzero8();
    float mrow[8], lrow[8];
    for (int i = 0; i < 8; ++i) { mrow[i] = -3.0e38f; lrow[i] = 0.0f; }

    const int kend = qblk * 128 + 128;
    const int ldr_kt = tid >> 3;
    const int ldr_d0 = (tid & 7) * 8;

    for (int kb0 = 0; kb0 < kend; kb0 += 32) {
        // Stage K tile (row-major) and V tile (transposed).
        {
            const size_t gsrc = headOff + (size_t)(kb0 + ldr_kt) * 64 + ldr_d0;
            const uint4 kv = *reinterpret_cast<const uint4*>(kb + gsrc);
            *reinterpret_cast<uint4*>(&Kt[ldr_kt * LKT + ldr_d0]) = kv;
            const uint4 vv = *reinterpret_cast<const uint4*>(vb + gsrc);
            const unsigned short* vs = reinterpret_cast<const unsigned short*>(&vv);
            for (int j = 0; j < 8; ++j)
                Vt[(ldr_d0 + j) * LVT + ldr_kt] = vs[j];
        }
        __syncthreads();

        // Scores S = Q * K^T (two 16-col halves).
        v8f s0 = vzero8(), s1 = vzero8();
        for (int j = 0; j < 2; ++j) {
            Frag kf0, kf1;
            const int kbd = 32 * j + (hi ? 8 : 0);
            const unsigned short* kr0 = &Kt[lc * LKT + kbd];
            kf0.q[0] = *reinterpret_cast<const uint4*>(kr0);
            kf0.q[1] = *reinterpret_cast<const uint4*>(kr0 + 16);
            const unsigned short* kr1 = &Kt[(16 + lc) * LKT + kbd];
            kf1.q[0] = *reinterpret_cast<const uint4*>(kr1);
            kf1.q[1] = *reinterpret_cast<const uint4*>(kr1 + 16);
            s0 = wmma_bf16(qf[j], kf0, s0);
            s1 = wmma_bf16(qf[j], kf1, s1);
        }

        // Online softmax + write P (bf16) to per-wave LDS scratch.
        const float scale = 0.125f;  // 1/sqrt(64)
        for (int i = 0; i < 8; ++i) {
            const int qt = qbase + i + 8 * hi;
            const int c0 = kb0 + lc, c1 = c0 + 16;
            const bool v0 = (c0 <= qt), v1 = (c1 <= qt);
            const float x0 = v0 ? s0[i] * scale : -3.0e38f;
            const float x1 = v1 ? s1[i] * scale : -3.0e38f;
            float mx = fmaxf(x0, x1);
            for (int off = 1; off < 16; off <<= 1)
                mx = fmaxf(mx, __shfl_xor(mx, off, 16));
            const float mnew = fmaxf(mrow[i], mx);
            const float fac = __expf(mrow[i] - mnew);
            const float e0 = v0 ? __expf(x0 - mnew) : 0.0f;
            const float e1 = v1 ? __expf(x1 - mnew) : 0.0f;
            float ls = e0 + e1;
            for (int off = 1; off < 16; off <<= 1)
                ls += __shfl_xor(ls, off, 16);
            lrow[i] = lrow[i] * fac + ls;
            mrow[i] = mnew;
            for (int c = 0; c < 4; ++c) o[c][i] *= fac;
            unsigned short* prow = &Pw[(wv * 16 + i + 8 * hi) * LPW];
            prow[lc] = f2bf(e0);
            prow[16 + lc] = f2bf(e1);
        }
        __syncthreads();

        // O += P(16x32) * V(32x64): P re-read as an A fragment, V from
        // transposed tile as B fragments.
        Frag pf;
        {
            const int klo = hi ? 8 : 0;
            const unsigned short* prow = &Pw[(wv * 16 + lc) * LPW];
            pf.q[0] = *reinterpret_cast<const uint4*>(prow + klo);
            pf.q[1] = *reinterpret_cast<const uint4*>(prow + 16 + klo);
        }
        for (int c = 0; c < 4; ++c) {
            Frag vf;
            const unsigned short* vr = &Vt[(16 * c + lc) * LVT + (hi ? 8 : 0)];
            vf.q[0] = *reinterpret_cast<const uint4*>(vr);
            vf.q[1] = *reinterpret_cast<const uint4*>(vr + 16);
            o[c] = wmma_bf16(pf, vf, o[c]);
        }
        __syncthreads();
    }

    // Epilogue: normalize and store bf16 into [B,T,D] layout.
    const int b = bh >> 5, h = bh & 31;
    for (int i = 0; i < 8; ++i) {
        const int t = qbase + i + 8 * hi;
        const float linv = 1.0f / lrow[i];
        for (int c = 0; c < 4; ++c) {
            const int col = h * 64 + 16 * c + lc;
            attn[((size_t)(b * 512 + t)) * 2048 + col] = f2bf(o[c][i] * linv);
        }
    }
}

// ---------------------------------------------------------------------------
// Host-side orchestration.
// ---------------------------------------------------------------------------
extern "C" void kernel_launch(void* const* d_in, const int* in_sizes, int n_in,
                              void* d_out, int out_size, void* d_ws, size_t ws_size,
                              hipStream_t stream) {
    (void)in_sizes; (void)n_in; (void)out_size; (void)ws_size;

    const float* x     = (const float*)d_in[0];
    const float* wq    = (const float*)d_in[1];
    const float* wk    = (const float*)d_in[2];
    const float* wv    = (const float*)d_in[3];
    const float* wo    = (const float*)d_in[4];
    const float* w1    = (const float*)d_in[5];
    const float* w2    = (const float*)d_in[6];
    const float* w3    = (const float*)d_in[7];
    const float* norm1 = (const float*)d_in[8];
    const float* norm2 = (const float*)d_in[9];
    const int*   pos   = (const int*)d_in[10];

    float* out  = (float*)d_out;            // x_out  [8,512,2048]
    float* kout = out + 8388608;            // new_k  [8,32,512,64]
    float* vout = out + 16777216;           // new_v  [8,32,512,64]

    // Workspace layout (byte offsets, buffers reused across phases).
    const size_t MB = 1024 * 1024;
    char* w8 = (char*)d_ws;
    unsigned short* nxA   = (unsigned short*)(w8 + 0);         // 16.8 MB (nx / nx2)
    float*          qpre  = (float*)        (w8 + 32 * MB);    // 33.6 MB
    float*          kpre  = (float*)        (w8 + 68 * MB);    // 33.6 MB
    unsigned short* q_bf  = (unsigned short*)(w8 + 104 * MB);  // 16.8 MB
    unsigned short* k_bf  = (unsigned short*)(w8 + 124 * MB);  // 16.8 MB
    unsigned short* v_bf  = (unsigned short*)(w8 + 144 * MB);  // 16.8 MB
    unsigned short* attnB = (unsigned short*)(w8 + 164 * MB);  // 16.8 MB
    unsigned short* h1    = (unsigned short*)(w8 + 32 * MB);   // 67.1 MB (reuse q/kpre)
    unsigned short* hb    = (unsigned short*)(w8 + 104 * MB);  // 67.1 MB (reuse q/k/v bf)

    const dim3 blk(256);
    const int M = 4096, D = 2048, F = 8192;

    // 1. RMSNorm -> bf16 nx
    rmsnorm_kernel<<<M, blk, 0, stream>>>(x, norm1, nxA);

    // 2. Q/K/V projections (WMMA)
    gemm_bf16_kernel<GM_F32><<<dim3(D / 128, M / 64), blk, 0, stream>>>(
        nxA, wq, M, D, D, qpre, nullptr, nullptr, nullptr);
    gemm_bf16_kernel<GM_F32><<<dim3(D / 128, M / 64), blk, 0, stream>>>(
        nxA, wk, M, D, D, kpre, nullptr, nullptr, nullptr);
    gemm_bf16_kernel<GM_HEADS_DUAL><<<dim3(D / 128, M / 64), blk, 0, stream>>>(
        nxA, wv, M, D, D, vout, v_bf, nullptr, nullptr);

    // 3. RoPE: heads-layout bf16 Q/K + fp32 new_k into d_out
    rope_kernel<<<(8 * 32 * 512 * 32) / 256, blk, 0, stream>>>(
        qpre, kpre, pos, q_bf, k_bf, kout);

    // 4. Causal flash attention (WMMA) -> bf16 attn in [B,T,D]
    attn_kernel<<<dim3(4, 256), blk, 0, stream>>>(q_bf, k_bf, v_bf, attnB);

    // 5. Output projection + residual -> d_out x region
    gemm_bf16_kernel<GM_RESID><<<dim3(D / 128, M / 64), blk, 0, stream>>>(
        attnB, wo, M, D, D, out, nullptr, x, nullptr);

    // 6. RMSNorm2 -> bf16 nx2 (reuses nxA)
    rmsnorm_kernel<<<M, blk, 0, stream>>>(out, norm2, nxA);

    // 7. FFN: silu(nx2 w1^T) * (nx2 w3^T) then w2^T with residual
    gemm_bf16_kernel<GM_SILU><<<dim3(F / 128, M / 64), blk, 0, stream>>>(
        nxA, w1, M, F, D, nullptr, h1, nullptr, nullptr);
    gemm_bf16_kernel<GM_MUL><<<dim3(F / 128, M / 64), blk, 0, stream>>>(
        nxA, w3, M, F, D, nullptr, hb, nullptr, h1);
    gemm_bf16_kernel<GM_RESID><<<dim3(D / 128, M / 64), blk, 0, stream>>>(
        hb, w2, M, D, F, out, nullptr, out, nullptr);
}
